// LampAttention_23175643529609
// MI455X (gfx1250) — matrix operand order repaired and checked by reading, not compiled, not run on hardware
//
#include <hip/hip_runtime.h>
#include <hip/hip_bf16.h>

// ---------------------------------------------------------------------------
// LAMP attention for gfx1250 (MI455X).
// M_BITS = 7 mantissa rounding == bf16, so every low-precision GEMM maps to
// v_wmma_f32_16x16x32_bf16 with the f32 accumulator re-quantized to bf16 per
// 64-deep K block (BLOCK_K=64). The high-precision rescore path uses the f32
// matrix op v_wmma_f32_16x16x4_f32. Data movement: Tensor Data Mover (async,
// TENSORcnt) with padded-row descriptors; LDS fragments read as ds_load_b128.
// ---------------------------------------------------------------------------

typedef __attribute__((ext_vector_type(16))) __bf16 v16bf;
typedef __attribute__((ext_vector_type(8)))  __bf16 v8bf;
typedef __attribute__((ext_vector_type(8)))  float  v8f;
typedef __attribute__((ext_vector_type(2)))  float  v2f;
typedef __attribute__((ext_vector_type(4)))  unsigned int u32x4;
typedef __attribute__((ext_vector_type(8)))  int i32x8;
typedef __attribute__((ext_vector_type(4)))  int i32x4;

#define INF __builtin_inff()

__device__ __forceinline__ v8f wmma_bf16(v16bf a, v16bf b, v8f c) {
  return __builtin_amdgcn_wmma_f32_16x16x32_bf16(false, a, false, b,
                                                 (short)0, c, false, false);
}
// round_mantissa(x,7) == bf16 RNE quantization
__device__ __forceinline__ float rm7(float x) { return (float)(__bf16)x; }

#if __has_builtin(__builtin_amdgcn_wmma_f32_16x16x4_f32)
#define HAVE_WMMA_F32 1
__device__ __forceinline__ v8f wmma_f32(v2f a, v2f b, v8f c) {
  // D = A(16x4 f32) * B(4x16 f32) + C(16x16 f32) -- full-precision path
  return __builtin_amdgcn_wmma_f32_16x16x4_f32(false, a, false, b,
                                               (short)0, c, false, false);
}
#else
#define HAVE_WMMA_F32 0
#endif

// ---------------------------------------------------------------------------
// Tensor Data Mover: async 2D f32 tile load Global -> LDS (CDNA5 ISA).
// D# group0: count=1 | lds_addr | global_addr[56:0] | type=2.
// D# group1: data_size=4B, pad_enable, pad_interval=64dw, pad_amount=4dw
//            (=> LDS row stride 68 floats), tensor_dim0=row_stride,
//            tensor_dim1=1<<20, tile_dim0=tile_w, tile_dim1=tile_h,
//            tensor_dim0_stride=row_stride.  Tracked by TENSORcnt.
// ---------------------------------------------------------------------------
__device__ __forceinline__ void tdm_load_tile_f32(unsigned lds_addr,
                                                  const float* gptr,
                                                  unsigned tile_w, unsigned tile_h,
                                                  unsigned row_stride)
{
  unsigned long long ga = (unsigned long long)(size_t)gptr;
  u32x4 g0;
  g0[0] = 1u;                                                  // count=1
  g0[1] = lds_addr;                                            // lds_addr
  g0[2] = (unsigned)ga;                                        // ga[31:0]
  g0[3] = (unsigned)((ga >> 32) & 0x1FFFFFFull) | (2u << 30);  // ga[56:32]|type=2
  i32x8 g1;
  g1[0] = (int)((2u << 16) | (1u << 20) | (5u << 22) | (3u << 25));
  g1[1] = (int)((row_stride & 0xFFFFu) << 16);   // tensor_dim0[15:0]
  g1[2] = (int)((row_stride >> 16) & 0xFFFFu);   // tensor_dim0[31:16], dim1 lo=0
  g1[3] = (int)(16u | (tile_w << 16));           // dim1 hi (=1<<20), tile_dim0
  g1[4] = (int)(tile_h & 0xFFFFu);               // tile_dim1, tile_dim2=0
  g1[5] = (int)row_stride;                       // tensor_dim0_stride lo32
  g1[6] = 0;
  g1[7] = 0;
  i32x4 gz = {0, 0, 0, 0};                       // 2D: groups 2/3 unused
  asm volatile("tensor_load_to_lds %0, %1, %2, %3"
               :: "s"(g0), "s"(g1), "s"(gz), "s"(gz)
               : "memory");
}

// ---------------------------------------------------------------------------
// Vectorized fragment loads (contiguous k = 16*hi + s, same in A and B). Per-lane data is
// contiguous: A frag and B frag ([n][k] tiles) = one 16-elt K run each.
// ---------------------------------------------------------------------------
__device__ __forceinline__ v16bf frag_a_bf16(const __bf16* base, int row,
                                             int k0, int hi, int ld) {
  const v8bf* p = (const v8bf*)(base + row * ld + k0 + 16 * hi);
  return __builtin_shufflevector(p[0], p[1],
      0,1,2,3,4,5,6,7,8,9,10,11,12,13,14,15);
}
__device__ __forceinline__ v16bf frag_b_bf16(const __bf16* base, int n,
                                             int k0, int hi, int ld) {
  const v8bf* p = (const v8bf*)(base + n * ld + k0 + 16 * hi);
  return __builtin_shufflevector(p[0], p[1],
      0,1,2,3,4,5,6,7,8,9,10,11,12,13,14,15);
}
__device__ __forceinline__ v16bf frag_a_f32(const float* base, int row,
                                            int k0, int hi, int ld) {
  const float* p = base + row * ld + k0 + 16 * hi;
  v16bf r;
#pragma unroll
  for (int i = 0; i < 8; ++i) { r[i] = (__bf16)p[i]; r[8+i] = (__bf16)p[8+i]; }
  return r;
}
__device__ __forceinline__ v16bf frag_b_f32(const float* base, int n,
                                            int k0, int hi, int ld) {
  const float* p = base + n * ld + k0 + 16 * hi;
  v16bf r;
#pragma unroll
  for (int i = 0; i < 16; ++i) r[i] = (__bf16)p[i];
  return r;
}

// ===========================================================================
// Kernel 1/3: blocked low-precision GEMM  out = rmacc_bf16(A[M,K] @ W[N,K]^T)+b
// 128 thr (4 waves), 16x64 macro tile. TDM double-buffers the K chunks:
// wave0 issues next A/W tiles, waits TENSORcnt<=2, block barrier, compute.
// ===========================================================================
__global__ __launch_bounds__(128) void gemm_lp_kernel(
    const float* __restrict__ A, const float* __restrict__ W,
    const float* __restrict__ bias, float* __restrict__ out,
    int M, int N, int K)
{
  __shared__ float Af[2][16][68];   // TDM-padded rows (64 data + 4 pad dwords)
  __shared__ float Wf[2][64][68];

  const int tid  = threadIdx.x;
  const int wave = tid >> 5, lane = tid & 31, lh = lane & 15, hi = lane >> 4;
  const int m0 = blockIdx.y * 16;
  const int n0 = blockIdx.x * 64;
  const int nC = K >> 6;

  if (wave == 0) {
    tdm_load_tile_f32((unsigned)(size_t)&Af[0][0][0], A + (size_t)m0 * K, 64, 16, (unsigned)K);
    tdm_load_tile_f32((unsigned)(size_t)&Wf[0][0][0], W + (size_t)n0 * K, 64, 64, (unsigned)K);
  }

  v8f tot = {0.f,0.f,0.f,0.f,0.f,0.f,0.f,0.f};

  for (int c = 0; c < nC; ++c) {
    const int buf = c & 1;
    if (wave == 0) {
      if (c + 1 < nC) {                         // prefetch next chunk via TDM
        const int k0 = (c + 1) << 6;
        tdm_load_tile_f32((unsigned)(size_t)&Af[buf^1][0][0],
                          A + (size_t)m0 * K + k0, 64, 16, (unsigned)K);
        tdm_load_tile_f32((unsigned)(size_t)&Wf[buf^1][0][0],
                          W + (size_t)n0 * K + k0, 64, 64, (unsigned)K);
        __builtin_amdgcn_s_wait_tensorcnt(2);   // current chunk arrived
      } else {
        __builtin_amdgcn_s_wait_tensorcnt(0);
      }
    }
    __syncthreads();

    const float* a  = &Af[buf][0][0];
    const float* wb = &Wf[buf][0][0];
    v16bf a0 = frag_a_f32(a, lh, 0,  hi, 68);
    v16bf a1 = frag_a_f32(a, lh, 32, hi, 68);
    v16bf b0 = frag_b_f32(wb, wave*16 + lh, 0,  hi, 68);
    v16bf b1 = frag_b_f32(wb, wave*16 + lh, 32, hi, 68);

    v8f part = {0.f,0.f,0.f,0.f,0.f,0.f,0.f,0.f};
    part = wmma_bf16(a0, b0, part);
    part = wmma_bf16(a1, b1, part);
#pragma unroll
    for (int e = 0; e < 8; ++e)                 // blocked lp accumulation
      tot[e] = rm7(tot[e] + rm7(part[e]));
    __syncthreads();
  }

#pragma unroll
  for (int e = 0; e < 8; ++e) {
    const int m = e + 8*hi;
    const int n = wave*16 + lh;
    out[(size_t)(m0 + m) * N + n0 + n] = tot[e] + bias[n0 + n];
  }
}

// ===========================================================================
// Kernel 2/3: fused LAMP attention per (b, h, 16-row tile). 256 thr (8 waves).
// Dynamic LDS ~186 KB (CDNA5 WGP has 320 KB); phase-disjoint regions unioned.
// ===========================================================================
#define T_SEQ  1024
#define HID    64
#define QKV_LD 3072

#define OFF_S     0                     // f32 [16][1024]                65536
#define OFF_U     65536                 // union: ATTf f32[16][1032] / ATTb bf16[16][1032]
#define OFF_QF    (OFF_U + 66048)       // f32 [16][64]                   4096
#define OFF_QB    (OFF_QF + 4096)       // bf16 [16][72]                  2304
#define OFF_KT    (OFF_QB + 2304)       // union: f32 slabs [8][16][68] / Vb bf16[64][72]
#define OFF_MK    (OFF_KT + 34816)      // u8 [16][1024]                 16384
#define OFF_RED   (OFF_MK + 16384)      // f32 [16][16]
#define OFF_MX    (OFF_RED + 1024)
#define OFF_INV   (OFF_MX + 64)
#define OFF_TH    (OFF_INV + 64)
#define ATTN_SMEM (OFF_TH + 64)         // 190400 bytes

__device__ void softmax_stats(const float* S, float* red, float* mx, float* inv,
                              int tid)
{
  const int r = tid >> 4, l = tid & 15;
  float m = -INF;
  for (int c = l; c < T_SEQ; c += 16) m = fmaxf(m, S[r*T_SEQ + c]);
  red[r*16 + l] = m;
  __syncthreads();
  if (l == 0) {
    float mm = -INF;
    for (int i = 0; i < 16; ++i) mm = fmaxf(mm, red[r*16 + i]);
    mx[r] = mm;
  }
  __syncthreads();
  const float mm = mx[r];
  float s = 0.f;
  for (int c = l; c < T_SEQ; c += 16) s += __expf(S[r*T_SEQ + c] - mm);
  red[r*16 + l] = s;
  __syncthreads();
  if (l == 0) {
    float ss = 0.f;
    for (int i = 0; i < 16; ++i) ss += red[r*16 + i];
    inv[r] = 1.f / ss;
  }
  __syncthreads();
}

__global__ __launch_bounds__(256) void attn_kernel(const float* __restrict__ qkv,
                                                   float* __restrict__ y)
{
  extern __shared__ char smem[];
  float*  S    = (float*) (smem + OFF_S);
  float*  ATTf = (float*) (smem + OFF_U);     // softmax#1 rows (mask phase)
  __bf16* ATTb = (__bf16*)(smem + OFF_U);     // softmax#2 rows (att@V phase)
  float*  Qf   = (float*) (smem + OFF_QF);
  __bf16* Qb   = (__bf16*)(smem + OFF_QB);
  float*  Kt   = (float*) (smem + OFF_KT);    // per-wave TDM K slabs (f32)
  __bf16* Vb   = (__bf16*)(smem + OFF_KT);    // V block [d][t] (att@V phase)
  unsigned char* Mk = (unsigned char*)(smem + OFF_MK);
  float*  red  = (float*) (smem + OFF_RED);
  float*  mx   = (float*) (smem + OFF_MX);
  float*  inv  = (float*) (smem + OFF_INV);
  float*  th   = (float*) (smem + OFF_TH);

  const int tid  = threadIdx.x;
  const int wave = tid >> 5, lane = tid & 31, lh = lane & 15, hi = lane >> 4;
  const int mt = blockIdx.x & 63;
  const int bh = blockIdx.x >> 6;
  const int h = bh & 15, b = bh >> 4;
  const int t0 = mt * 16;
  const float scale = 0.125f;                 // 1/sqrt(64)

  const float* q_base = qkv + (size_t)b * T_SEQ * QKV_LD + h * HID;
  const float* k_base = q_base + 1024;
  const float* v_base = q_base + 2048;

  // ---- stage Q (f32 for hp path; bf16, stride 72, for WMMA fragments)
  for (int e = tid; e < 16*HID; e += 256) {
    const int r = e >> 6, d = e & 63;
    const float v = q_base[(size_t)(t0 + r) * QKV_LD + d];
    Qf[r*HID + d] = v;
    Qb[r*72 + d]  = (__bf16)v;
  }
  __syncthreads();

  float* ktw = Kt + wave * (16*68);           // wave-private TDM slab

  // ---- phase 1: lp scores via bf16 WMMA; K tiles arrive by TDM
  for (int it = 0; it < 8; ++it) {
    const int ct = wave + it*8;
    const int c0 = ct * 16;
    const bool any = (c0 <= t0 + 15);         // wave-uniform
    v8f acc = {0.f,0.f,0.f,0.f,0.f,0.f,0.f,0.f};
    if (any) {
      asm volatile("s_wait_dscnt 0" ::: "memory");   // slab reuse safety
      tdm_load_tile_f32((unsigned)(size_t)ktw,
                        k_base + (size_t)c0 * QKV_LD, 64, 16, QKV_LD);
      __builtin_amdgcn_s_wait_tensorcnt(0);
      v16bf a0 = frag_a_bf16(Qb, lh, 0,  hi, 72);
      v16bf a1 = frag_a_bf16(Qb, lh, 32, hi, 72);
      v16bf b0 = frag_b_f32(ktw, lh, 0,  hi, 68);
      v16bf b1 = frag_b_f32(ktw, lh, 32, hi, 68);
      acc = wmma_bf16(a0, b0, acc);
      acc = wmma_bf16(a1, b1, acc);
    }
#pragma unroll
    for (int e = 0; e < 8; ++e) {
      const int m = e + 8*hi;
      const int c = c0 + lh;
      S[m*T_SEQ + c] = (any && c <= t0 + m) ? rm7(rm7(acc[e]) * scale) : -INF;
    }
  }
  __syncthreads();

  // ---- softmax #1
  softmax_stats(S, red, mx, inv, tid);

  // ---- materialize att#1 rows once (kills the O(T^2) exp recompute)
  for (int e = tid; e < 16*T_SEQ; e += 256) {
    const int r = e >> 10, c = e & (T_SEQ - 1);
    ATTf[r*1032 + c] = __expf(S[r*T_SEQ + c] - mx[r]) * inv[r];
  }
  __syncthreads();

  // ---- LAMP threshold per row: th = (2-TAU) - 2*att_min = 1 - 2*att_min
  {
    const int r = tid >> 4, l = tid & 15;
    float mn = INF;
    for (int c = l; c < T_SEQ; c += 16) mn = fminf(mn, ATTf[r*1032 + c]);
    red[r*16 + l] = mn;
    __syncthreads();
    if (l == 0) {
      float m2 = INF;
      for (int i = 0; i < 16; ++i) m2 = fminf(m2, red[r*16 + i]);
      th[r] = 1.0f - 2.0f * m2;
    }
    __syncthreads();
  }

  // ---- LAMP mask: keep(j) <=> stable-descending exclusive prefix(j) < th
  for (int e = tid; e < 16*T_SEQ; e += 256) {
    const int r = e >> 10, c = e & (T_SEQ - 1);
    const float aj = ATTf[r*1032 + c];
    float g = 0.f;
    for (int k = 0; k < T_SEQ; ++k) {
      const float ak = ATTf[r*1032 + k];
      if (ak > aj || (ak == aj && k < c)) g += ak;
    }
    Mk[e] = (g < th[r]) ? 1 : 0;
  }
  __syncthreads();

  // ---- high-precision rescore where kept: f32 WMMA (16x16x4), K tiles by TDM
  for (int it = 0; it < 8; ++it) {
    const int ct = wave + it*8;
    const int c0 = ct * 16;
    if (c0 > t0 + 15) continue;
    asm volatile("s_wait_dscnt 0" ::: "memory");
    tdm_load_tile_f32((unsigned)(size_t)ktw,
                      k_base + (size_t)c0 * QKV_LD, 64, 16, QKV_LD);
    __builtin_amdgcn_s_wait_tensorcnt(0);
#if HAVE_WMMA_F32
    v8f hp = {0.f,0.f,0.f,0.f,0.f,0.f,0.f,0.f};
#pragma unroll
    for (int ks = 0; ks < HID; ks += 4) {       // 16 chained f32 WMMAs
      v2f a, bq;
      a[0]  = Qf[lh*HID + ks + 2*hi];           // A 16x4: lane M=lh, K=2*hi(+1)
      a[1]  = Qf[lh*HID + ks + 2*hi + 1];
      bq[0] = ktw[lh*68 + ks + 2*hi];           // B 4x16: lane N=lh, K=2*hi(+1)
      bq[1] = ktw[lh*68 + ks + 2*hi + 1];
      hp = wmma_f32(a, bq, hp);
    }
#pragma unroll
    for (int e = 0; e < 8; ++e) {
      const int m = e + 8*hi;
      const int c = c0 + lh;
      if (c <= t0 + m && Mk[m*T_SEQ + c])
        S[m*T_SEQ + c] = hp[e] * scale;
    }
#else
#pragma unroll
    for (int e = 0; e < 8; ++e) {
      const int m = e + 8*hi;
      const int c = c0 + lh;
      if (c <= t0 + m && Mk[m*T_SEQ + c]) {
        float acc = 0.f;
#pragma unroll
        for (int k = 0; k < HID; ++k)
          acc = fmaf(Qf[m*HID + k], ktw[lh*68 + k], acc);
        S[m*T_SEQ + c] = acc * scale;
      }
    }
#endif
  }
  __syncthreads();

  // ---- softmax #2 on updated scores
  softmax_stats(S, red, mx, inv, tid);

  // ---- materialize att#2 as bf16 A-rows (stride 1032, b128-loadable)
  for (int e = tid; e < 16*T_SEQ; e += 256) {
    const int r = e >> 10, c = e & (T_SEQ - 1);
    ATTb[r*1032 + c] = (__bf16)(__expf(S[r*T_SEQ + c] - mx[r]) * inv[r]);
  }
  __syncthreads();

  // ---- y = att @ V : bf16 WMMA, blocked lp accumulation over K=1024
  v8f tot = {0.f,0.f,0.f,0.f,0.f,0.f,0.f,0.f};
  for (int kb0 = 0; kb0 < T_SEQ; kb0 += 64) {
    __syncthreads();
    // all 8 waves stage V block (64 t x 64 d) transposed -> Vb[d][t] bf16
    for (int e = tid; e < 64*64; e += 256) {
      const int d = e & 63, t = e >> 6;
      Vb[d*72 + t] = (__bf16)v_base[(size_t)(kb0 + t) * QKV_LD + d];
    }
    __syncthreads();
    if (wave < 4) {                           // 4 waves cover hd = 4 x 16
      v16bf a0 = frag_a_bf16(ATTb, lh, kb0,      hi, 1032);
      v16bf a1 = frag_a_bf16(ATTb, lh, kb0 + 32, hi, 1032);
      v16bf b0 = frag_b_bf16(Vb, wave*16 + lh, 0,  hi, 72);
      v16bf b1 = frag_b_bf16(Vb, wave*16 + lh, 32, hi, 72);
      v8f part = {0.f,0.f,0.f,0.f,0.f,0.f,0.f,0.f};
      part = wmma_bf16(a0, b0, part);
      part = wmma_bf16(a1, b1, part);
#pragma unroll
      for (int e = 0; e < 8; ++e)
        tot[e] = rm7(tot[e] + rm7(part[e]));
    }
  }
  if (wave < 4) {
#pragma unroll
    for (int e = 0; e < 8; ++e) {
      const int m = e + 8*hi;
      y[(size_t)(b*T_SEQ + t0 + m) * 1024 + h*HID + wave*16 + lh] = tot[e];
    }
  }
}

// ===========================================================================
// Launch: QKV gemm -> fused LAMP attention -> output projection
// Workspace: qkv f32 [2048x3072] (25.2 MB) + y f32 [2048x1024] (8.4 MB)
// ===========================================================================
extern "C" void kernel_launch(void* const* d_in, const int* in_sizes, int n_in,
                              void* d_out, int out_size, void* d_ws, size_t ws_size,
                              hipStream_t stream)
{
  (void)in_sizes; (void)n_in; (void)out_size; (void)ws_size;
  const float* x  = (const float*)d_in[0];   // [2,1024,1024]
  const float* Wa = (const float*)d_in[1];   // [3072,1024]
  const float* ba = (const float*)d_in[2];   // [3072]
  const float* Wp = (const float*)d_in[3];   // [1024,1024]
  const float* bp = (const float*)d_in[4];   // [1024]
  float* out = (float*)d_out;                // [2,1024,1024]

  float* qkv = (float*)d_ws;                     // [2048 x 3072]
  float* yb  = qkv + (size_t)2048 * 3072;        // [2048 x 1024]

  gemm_lp_kernel<<<dim3(3072/64, 2048/16), 128, 0, stream>>>(
      x, Wa, ba, qkv, 2048, 3072, 1024);

  attn_kernel<<<dim3(2*16*64), 256, ATTN_SMEM, stream>>>(qkv, yb);

  gemm_lp_kernel<<<dim3(1024/64, 2048/16), 128, 0, stream>>>(
      yb, Wp, bp, out, 2048, 1024, 1024);
}
